// GCN_Net_47253230191022
// MI455X (gfx1250) — matrix operand order repaired and checked by reading, not compiled, not run on hardware
//
#include <hip/hip_runtime.h>
#include <stdint.h>

typedef __attribute__((ext_vector_type(16))) __bf16 v16bf;
typedef __attribute__((ext_vector_type(8)))  __bf16 v8bf;
typedef __attribute__((ext_vector_type(8)))  float  v8f;

#define D_IN  300
#define K1P   320   // D_IN padded to multiple of 32 for bf16 WMMA K-steps
#define D_HID 512   // already multiple of 32
#define N2P   304   // layer-2 output cols padded to multiple of 16 (19 tiles)
#define MAX_KP 512

// gfx1250 async global->LDS copy (ASYNCcnt) via inline asm; flip to 0 to fall
// back to load + ds_store if the assembler rejects the mnemonic.
#define USE_ASYNC_LDS 1

static __device__ __forceinline__ unsigned short f32_to_bf16(float f) {
  union { float f; uint32_t u; } c; c.f = f;
  uint32_t r = c.u + 0x7FFFu + ((c.u >> 16) & 1u);  // round-to-nearest-even
  return (unsigned short)(r >> 16);
}

// ---------------- degree / normalization ----------------
__global__ void k_deg_init(float* deg, int N) {
  int i = blockIdx.x * blockDim.x + threadIdx.x;
  if (i < N) deg[i] = 1.0f;                       // self-loop contribution
}

__global__ void k_deg_acc(const long long* __restrict__ dst, float* deg, int E) {
  int e = blockIdx.x * blockDim.x + threadIdx.x;
  if (e < E) atomicAdd(&deg[(int)dst[e]], 1.0f);
}

__global__ void k_dinv(float* deg, int N) {
  int i = blockIdx.x * blockDim.x + threadIdx.x;
  if (i < N) deg[i] = rsqrtf(deg[i]);             // deg >= 1 always
}

// ---------------- fp32 -> bf16 conversions (with padding/transpose) ----------------
__global__ void k_cvt_x(const float* __restrict__ x, unsigned short* __restrict__ xb) {
  int i = blockIdx.x;                              // node
  for (int k = threadIdx.x; k < K1P; k += blockDim.x)
    xb[(size_t)i * K1P + k] = (k < D_IN) ? f32_to_bf16(x[(size_t)i * D_IN + k]) : (unsigned short)0;
}

// W1:[300,512] row-major -> w1t:[512,320] (B^T, K padded with zeros)
__global__ void k_cvt_w1t(const float* __restrict__ W1, unsigned short* __restrict__ w1t) {
  int n = blockIdx.x;                              // 0..511
  for (int k = threadIdx.x; k < K1P; k += blockDim.x)
    w1t[(size_t)n * K1P + k] = (k < D_IN) ? f32_to_bf16(W1[(size_t)k * D_HID + n]) : (unsigned short)0;
}

// W2:[512,300] row-major -> w2t:[304,512] (B^T, N padded with zero rows)
__global__ void k_cvt_w2t(const float* __restrict__ W2, unsigned short* __restrict__ w2t) {
  int n = blockIdx.x;                              // 0..303
  for (int k = threadIdx.x; k < D_HID; k += blockDim.x)
    w2t[(size_t)n * D_HID + k] = (n < D_IN) ? f32_to_bf16(W2[(size_t)k * D_IN + n]) : (unsigned short)0;
}

// ---------------- bf16 WMMA GEMM: C[M,Ncols] = A[M,Kp] * Bt[Ncols,Kp]^T ----------------
// Block = 256 threads = 8 waves; block stages the 16 x Kp A slab into LDS once
// (async global->LDS on CDNA5), then each wave computes one 16x16 C tile with
// A fragments from LDS (ds_load_b128) and B fragments from L2-resident global.
// Fragment layouts per CDNA5 ISA 7.12.2:
//   A 16x32 bf16: lane L holds row M=L%16; Ks {h*8..h*8+7, 16+h*8..} with h=L/16 (two 16B loads)
//   B 32x16 bf16: lane L holds col N=L%16; Ks {h*8..h*8+7, 16+h*8..}     (two 16B loads)
//   C 16x16 f32 : lane L holds col N=L%16; VGPR r holds row M = r + 8*(L/16)
__global__ __launch_bounds__(256) void k_gemm_bf16(
    const unsigned short* __restrict__ A,
    const unsigned short* __restrict__ Bt,
    float* __restrict__ C,
    int Kp, int Ncols, int Ntiles)
{
  __shared__ unsigned short sA[16 * MAX_KP];       // 16 KB max

  const int wave = threadIdx.x >> 5;
  const int lane = threadIdx.x & 31;
  const int m0   = blockIdx.y * 16;

  // ---- cooperative stage: 16 x Kp bf16 slab, 16B chunks ----
  {
    const int chunks_per_row = Kp >> 3;            // 8 bf16 per 16B chunk
    const int total = 16 * chunks_per_row;
    for (int idx = threadIdx.x; idx < total; idx += 256) {
      const int r = idx / chunks_per_row;
      const int c = (idx - r * chunks_per_row) << 3;
#if USE_ASYNC_LDS
      unsigned laddr = (unsigned)(uintptr_t)(&sA[r * Kp + c]);                 // LDS byte offset (addr[31:0])
      unsigned long long gaddr =
          (unsigned long long)(uintptr_t)(A + (size_t)(m0 + r) * Kp + c);
      asm volatile("global_load_async_to_lds_b128 %0, %1, off"
                   :: "v"(laddr), "v"(gaddr)
                   : "memory");
#else
      *(uint4*)(&sA[r * Kp + c]) = *(const uint4*)(A + (size_t)(m0 + r) * Kp + c);
#endif
    }
  }
#if USE_ASYNC_LDS
  asm volatile("s_wait_asynccnt 0x0" ::: "memory");
#endif
  __syncthreads();

  const int tile = blockIdx.x * 8 + wave;
  if (tile >= Ntiles) return;                      // only barrier above; safe
  const int n0   = tile * 16;
  const int half = lane >> 4;   // 0|1
  const int sub  = lane & 15;

  const unsigned short* arow = sA + sub * Kp;                  // LDS
  const unsigned short* brow = Bt + (size_t)(n0 + sub) * Kp;   // global (L2-resident)

  v8f acc = {0.f, 0.f, 0.f, 0.f, 0.f, 0.f, 0.f, 0.f};

  for (int k0 = 0; k0 < Kp; k0 += 32) {
    const v8bf a_lo = *(const v8bf*)(arow + k0 + half * 8);
    const v8bf a_hi = *(const v8bf*)(arow + k0 + 16 + half * 8);
    const v8bf b_lo = *(const v8bf*)(brow + k0 + half * 8);
    const v8bf b_hi = *(const v8bf*)(brow + k0 + 16 + half * 8);
    v16bf a, b;
#pragma unroll
    for (int i = 0; i < 8; ++i) {
      a[i] = a_lo[i]; a[i + 8] = a_hi[i];
      b[i] = b_lo[i]; b[i + 8] = b_hi[i];
    }
    acc = __builtin_amdgcn_wmma_f32_16x16x32_bf16(
        /*neg_a=*/false, a, /*neg_b=*/false, b,
        /*c_mod=*/(short)0, acc, /*reuse_a=*/false, /*reuse_b=*/false);
  }

#pragma unroll
  for (int r = 0; r < 8; ++r)
    C[(size_t)(m0 + half * 8 + r) * Ncols + n0 + sub] = acc[r];
}

// ---------------- edge aggregation (L2-resident scatter) ----------------
__global__ void k_agg_init(const float* __restrict__ pre, const float* __restrict__ dinv,
                           float* __restrict__ agg, int F) {
  int i = blockIdx.x;
  float nrm = dinv[i] * dinv[i];                   // self-loop norm
  for (int j = threadIdx.x; j < F; j += blockDim.x)
    agg[(size_t)i * F + j] = pre[(size_t)i * F + j] * nrm;
}

#define EPB 4   // edges per block
__global__ void k_agg_edges(const long long* __restrict__ src, const long long* __restrict__ dst,
                            const float* __restrict__ dinv, const float* __restrict__ pre,
                            float* __restrict__ agg, int E, int F) {
  const int base = blockIdx.x * EPB;
  for (int t = 0; t < EPB; ++t) {
    const int e = base + t;
    if (e >= E) return;
    const int s = (int)src[e], d = (int)dst[e];
    // prefetch next edge's source row while this edge's atomics drain
    if (t + 1 < EPB && e + 1 < E) {
      const float* pn = pre + (size_t)(int)src[e + 1] * F;
      for (int j = threadIdx.x; j < F; j += blockDim.x)
        __builtin_prefetch(pn + j, 0, 0);          // global_prefetch_b8
    }
    const float nrm = dinv[s] * dinv[d];
    const float* pr = pre + (size_t)s * F;
    float*       ag = agg + (size_t)d * F;
    for (int j = threadIdx.x; j < F; j += blockDim.x)
      atomicAdd(&ag[j], pr[j] * nrm);
  }
}

// bias + ReLU, re-quantize to bf16 for the next GEMM's A matrix
__global__ void k_bias_relu_bf16(const float* __restrict__ agg, const float* __restrict__ bias,
                                 unsigned short* __restrict__ out, int F) {
  int i = blockIdx.x;
  for (int j = threadIdx.x; j < F; j += blockDim.x) {
    float v = agg[(size_t)i * F + j] + bias[j];
    out[(size_t)i * F + j] = f32_to_bf16(v > 0.f ? v : 0.f);
  }
}

// bias + ReLU into final fp32 output (strip N-padding 304 -> 300)
__global__ void k_bias_relu_out(const float* __restrict__ agg, const float* __restrict__ bias,
                                float* __restrict__ out, int Fs, int F) {
  int i = blockIdx.x;
  for (int j = threadIdx.x; j < F; j += blockDim.x) {
    float v = agg[(size_t)i * Fs + j] + bias[j];
    out[(size_t)i * F + j] = v > 0.f ? v : 0.f;
  }
}

extern "C" void kernel_launch(void* const* d_in, const int* in_sizes, int n_in,
                              void* d_out, int out_size, void* d_ws, size_t ws_size,
                              hipStream_t stream) {
  const float*     x   = (const float*)d_in[0];
  const long long* ei  = (const long long*)d_in[1];   // int64 [2,E]
  const float*     W1  = (const float*)d_in[2];
  const float*     b1  = (const float*)d_in[3];
  const float*     W2  = (const float*)d_in[4];
  const float*     b2  = (const float*)d_in[5];

  const int N = in_sizes[0] / D_IN;                   // 50000 (multiple of 16)
  const int E = in_sizes[1] / 2;                      // 400000
  const long long* srcI = ei;
  const long long* dstI = ei + E;

  // ---- workspace carve-up ----
  char*  ws  = (char*)d_ws;
  size_t off = 0;
  auto alloc = [&](size_t bytes) -> void* {
    void* p = ws + off;
    off = (off + bytes + 255) & ~(size_t)255;
    return p;
  };
  float*          dinv = (float*)         alloc((size_t)N * 4);
  unsigned short* xb   = (unsigned short*)alloc((size_t)N * K1P * 2);
  unsigned short* w1t  = (unsigned short*)alloc((size_t)D_HID * K1P * 2);
  unsigned short* w2t  = (unsigned short*)alloc((size_t)N2P * D_HID * 2);
  unsigned short* hb   = (unsigned short*)alloc((size_t)N * D_HID * 2);
  float*          pre  = (float*)         alloc((size_t)N * D_HID * 4);  // reused for layer-2 (N x 304)
  float*          agg  = (float*)         alloc((size_t)N * D_HID * 4);  // reused for layer-2 (N x 304)
  float*          out  = (float*)d_out;

  const int T = 256;

  // normalization
  k_deg_init<<<(N + T - 1) / T, T, 0, stream>>>(dinv, N);
  k_deg_acc <<<(E + T - 1) / T, T, 0, stream>>>(dstI, dinv, E);
  k_dinv    <<<(N + T - 1) / T, T, 0, stream>>>(dinv, N);

  // bf16 conversions
  k_cvt_x  <<<N,     128, 0, stream>>>(x,  xb);
  k_cvt_w1t<<<D_HID, 128, 0, stream>>>(W1, w1t);
  k_cvt_w2t<<<N2P,   128, 0, stream>>>(W2, w2t);

  // ---- layer 1: pre = x @ W1 (N x 512) ----
  {
    const int Ntiles = D_HID / 16;                    // 32
    dim3 grid((Ntiles + 7) / 8, N / 16);
    k_gemm_bf16<<<grid, 256, 0, stream>>>(xb, w1t, pre, K1P, D_HID, Ntiles);
  }
  k_agg_init <<<N, 256, 0, stream>>>(pre, dinv, agg, D_HID);
  k_agg_edges<<<(E + EPB - 1) / EPB, 256, 0, stream>>>(srcI, dstI, dinv, pre, agg, E, D_HID);
  k_bias_relu_bf16<<<N, 256, 0, stream>>>(agg, b1, hb, D_HID);

  // ---- layer 2: pre = h @ W2 (N x 304, cols 300..303 are zero-weight pad) ----
  {
    const int Ntiles = N2P / 16;                      // 19
    dim3 grid((Ntiles + 7) / 8, N / 16);
    k_gemm_bf16<<<grid, 256, 0, stream>>>(hb, w2t, pre, D_HID, N2P, Ntiles);
  }
  k_agg_init <<<N, 256, 0, stream>>>(pre, dinv, agg, N2P);
  k_agg_edges<<<(E + EPB - 1) / EPB, 256, 0, stream>>>(srcI, dstI, dinv, pre, agg, E, N2P);
  k_bias_relu_out<<<N, 256, 0, stream>>>(agg, b2, out, N2P, D_IN);
}
